// ParallelDroplessMLP_11793980195163
// MI455X (gfx1250) — matrix operand order repaired and checked by reading, not compiled, not run on hardware
//
#include <hip/hip_runtime.h>
#include <hip/hip_bf16.h>
#include <stdint.h>

#define N_TOK  4096
#define TOPK   2
#define NEXP   8
#define DMODEL 1024
#define HDIM   1024
#define NKROWS (N_TOK * TOPK)   // 8192

typedef __attribute__((ext_vector_type(16))) __bf16   v16bf;
typedef __attribute__((ext_vector_type(8)))  float    v8f;
typedef __attribute__((ext_vector_type(4)))  unsigned uint4v;
typedef __attribute__((ext_vector_type(2)))  unsigned uint2v;
typedef __attribute__((ext_vector_type(4)))  float    float4v;

union Frag { uint4v u[2]; v16bf v; };

__device__ __forceinline__ unsigned short f2bfu(float f) {
  unsigned u = __float_as_uint(f);
  u += 0x7FFFu + ((u >> 16) & 1u);           // round-to-nearest-even
  return (unsigned short)(u >> 16);
}

// CDNA5 async copy: global -> LDS, tracked by ASYNCcnt (CDNA5 ISA)
__device__ __forceinline__ void async_ld_b128(unsigned lds_addr, const void* gaddr) {
  asm volatile("global_load_async_to_lds_b128 %0, %1, off"
               :: "v"(lds_addr), "v"((unsigned long long)(size_t)gaddr)
               : "memory");
}
#define WAIT_ASYNC(n) asm volatile("s_wait_asynccnt " #n ::: "memory")

// ---------------------------------------------------------------- permutation
__global__ void moe_zero(int* __restrict__ cnt) {
  if (threadIdx.x < NEXP) cnt[threadIdx.x] = 0;
}

__global__ void moe_perm(const int* __restrict__ eidx,
                         const int* __restrict__ bspe,
                         int* __restrict__ cnt,
                         int* __restrict__ row_of,
                         int* __restrict__ src_of) {
  int i = blockIdx.x * blockDim.x + threadIdx.x;
  if (i >= NKROWS) return;
  int e = eidx[i];
  int off = 0;
#pragma unroll
  for (int j = 0; j < NEXP; ++j) off += (j < e) ? bspe[j] : 0;
  int pos = off + atomicAdd(&cnt[e], 1);
  row_of[i]   = pos;
  src_of[pos] = i;
}

// --------------------------------------------- fp32 -> bf16 transpose weights
// All three weight tensors are E x 1024 x 1024; output layout is [e][col][row]
// so GEMM B-tiles load row-major contiguous, same pattern as A-tiles.
__global__ __launch_bounds__(256) void moe_wcvt(
    const float* __restrict__ w1, const float* __restrict__ w2,
    const float* __restrict__ w3,
    unsigned short* __restrict__ w1t, unsigned short* __restrict__ w2t,
    unsigned short* __restrict__ w3t) {
  __shared__ float tile[32][33];
  int z = blockIdx.z;
  const float* src; unsigned short* dst;
  if (z < NEXP)            { src = w1 + ((size_t)z << 20);           dst = w1t + ((size_t)z << 20); }
  else if (z < 2 * NEXP)   { src = w3 + ((size_t)(z - NEXP) << 20);  dst = w3t + ((size_t)(z - NEXP) << 20); }
  else                     { src = w2 + ((size_t)(z - 2*NEXP) << 20);dst = w2t + ((size_t)(z - 2*NEXP) << 20); }
  int tx = threadIdx.x & 31, ty = threadIdx.x >> 5;   // 32 x 8
  int c0 = blockIdx.x * 32, r0 = blockIdx.y * 32;
#pragma unroll
  for (int j = 0; j < 4; ++j)
    tile[ty + j * 8][tx] = src[(size_t)(r0 + ty + j * 8) * 1024 + c0 + tx];
  __syncthreads();
#pragma unroll
  for (int j = 0; j < 4; ++j)
    dst[(size_t)(c0 + ty + j * 8) * 1024 + r0 + tx] = f2bfu(tile[tx][ty + j * 8]);
}

// ------------------------------------------------- gather x_perm, cvt to bf16
__global__ __launch_bounds__(256) void moe_gather_x(
    const float* __restrict__ x, const int* __restrict__ src_of,
    unsigned short* __restrict__ xb) {
  int p = blockIdx.x;
  int t = threadIdx.x;                       // 256 threads * 4 floats = 1024
  int tok = src_of[p] / TOPK;
  float4v v = reinterpret_cast<const float4v*>(x + (size_t)tok * DMODEL)[t];
  uint2v o;
  o.x = (unsigned)f2bfu(v.x) | ((unsigned)f2bfu(v.y) << 16);
  o.y = (unsigned)f2bfu(v.z) | ((unsigned)f2bfu(v.w) << 16);
  reinterpret_cast<uint2v*>(xb + (size_t)p * DMODEL)[t] = o;
}

// ------------------------------------- GEMM1: h = silu(x@w1) * (x@w3)  (bf16)
// Block tile 128(M) x 64(N), BK=32, 4x2 wave grid, 32x32 per wave.
// Double-buffered LDS filled by GLOBAL_LOAD_ASYNC_TO_LDS_B128 (ASYNCcnt),
// K-loop unrolled by 2 with compile-time buffer ids, tail peeled (branch-free body).
__global__ __launch_bounds__(256) void moe_gemm1(
    const unsigned short* __restrict__ xb,
    const unsigned short* __restrict__ w1t,
    const unsigned short* __restrict__ w3t,
    unsigned short* __restrict__ hb,
    const int* __restrict__ bspe) {
  __shared__ unsigned short As[2][128][40];
  __shared__ unsigned short B1[2][64][40];
  __shared__ unsigned short B3[2][64][40];

  const int tilesM = NKROWS / 128;           // 64
  int e  = blockIdx.x / tilesM;
  int mt = blockIdx.x - e * tilesM;
  int off = 0, gsz = 0;
#pragma unroll
  for (int j = 0; j < NEXP; ++j) { int b = bspe[j]; if (j < e) off += b; if (j == e) gsz = b; }
  if (mt * 128 >= gsz) return;               // uniform per block
  int row_start = off + mt * 128;
  int valid = gsz - mt * 128; if (valid > 128) valid = 128;
  int n0 = blockIdx.y * 64;

  int t = threadIdx.x;
  // A fill: thread t loads row t>>1, 16 elems at (t&1)*16 (two b128s)
  int rowA = t >> 1;
  int chA  = (t & 1) * 16;
  int arow = row_start + rowA; if (arow >= NKROWS) arow = NKROWS - 1;
  const unsigned short* agA = xb + (size_t)arow * DMODEL + chA;
  // B fill: thread t loads row t>>2, 8 elems at (t&3)*8 (one b128 per weight)
  int rowB = t >> 2;
  int chB  = (t & 3) * 8;
  const unsigned short* b1g = w1t + ((size_t)e << 20) + (size_t)(n0 + rowB) * DMODEL + chB;
  const unsigned short* b3g = w3t + ((size_t)e << 20) + (size_t)(n0 + rowB) * DMODEL + chB;

  int lane = t & 31;
  int wv   = t >> 5;
  int wm   = wv & 3, wn = wv >> 2;           // 4(M) x 2(N) waves, 32x32 each
  int half = lane >> 4;
  int l15  = lane & 15;
  int m0 = wm * 32 + l15, m1 = m0 + 16;
  int nA = wn * 32 + l15, nB = nA + 16;

  v8f c100 = {}, c101 = {}, c110 = {}, c111 = {};
  v8f c300 = {}, c301 = {}, c310 = {}, c311 = {};

  auto issue = [&](int buf, int kk) {
    async_ld_b128((unsigned)(size_t)&As[buf][rowA][chA],     agA + kk);
    async_ld_b128((unsigned)(size_t)&As[buf][rowA][chA + 8], agA + kk + 8);
    async_ld_b128((unsigned)(size_t)&B1[buf][rowB][chB],     b1g + kk);
    async_ld_b128((unsigned)(size_t)&B3[buf][rowB][chB],     b3g + kk);
  };

  auto compute = [&](int buf) {
    Frag a0, a1, p0, p1, q0, q1;
    // A frag: lane<16 -> K {0..7,16..23}; lane>=16 -> K {8..15,24..31}
    a0.u[0] = *reinterpret_cast<const uint4v*>(&As[buf][m0][half * 8]);
    a0.u[1] = *reinterpret_cast<const uint4v*>(&As[buf][m0][half * 8 + 16]);
    a1.u[0] = *reinterpret_cast<const uint4v*>(&As[buf][m1][half * 8]);
    a1.u[1] = *reinterpret_cast<const uint4v*>(&As[buf][m1][half * 8 + 16]);
    // B frag: lane<16 -> K {0..7,16..23}; lane>=16 -> K {8..15,24..31}
    p0.u[0] = *reinterpret_cast<const uint4v*>(&B1[buf][nA][half * 8]);
    p0.u[1] = *reinterpret_cast<const uint4v*>(&B1[buf][nA][half * 8 + 16]);
    p1.u[0] = *reinterpret_cast<const uint4v*>(&B1[buf][nB][half * 8]);
    p1.u[1] = *reinterpret_cast<const uint4v*>(&B1[buf][nB][half * 8 + 16]);
    q0.u[0] = *reinterpret_cast<const uint4v*>(&B3[buf][nA][half * 8]);
    q0.u[1] = *reinterpret_cast<const uint4v*>(&B3[buf][nA][half * 8 + 16]);
    q1.u[0] = *reinterpret_cast<const uint4v*>(&B3[buf][nB][half * 8]);
    q1.u[1] = *reinterpret_cast<const uint4v*>(&B3[buf][nB][half * 8 + 16]);

    c100 = __builtin_amdgcn_wmma_f32_16x16x32_bf16(false, a0.v, false, p0.v, (short)0, c100, false, false);
    c101 = __builtin_amdgcn_wmma_f32_16x16x32_bf16(false, a0.v, false, p1.v, (short)0, c101, false, false);
    c110 = __builtin_amdgcn_wmma_f32_16x16x32_bf16(false, a1.v, false, p0.v, (short)0, c110, false, false);
    c111 = __builtin_amdgcn_wmma_f32_16x16x32_bf16(false, a1.v, false, p1.v, (short)0, c111, false, false);
    c300 = __builtin_amdgcn_wmma_f32_16x16x32_bf16(false, a0.v, false, q0.v, (short)0, c300, false, false);
    c301 = __builtin_amdgcn_wmma_f32_16x16x32_bf16(false, a0.v, false, q1.v, (short)0, c301, false, false);
    c310 = __builtin_amdgcn_wmma_f32_16x16x32_bf16(false, a1.v, false, q0.v, (short)0, c310, false, false);
    c311 = __builtin_amdgcn_wmma_f32_16x16x32_bf16(false, a1.v, false, q1.v, (short)0, c311, false, false);
  };

  issue(0, 0);
  issue(1, 32);
  // steady state: 15 iterations, branch-free body; computes steps kk, kk+32
  for (int kk = 0; kk + 96 < DMODEL; kk += 64) {
    WAIT_ASYNC(0x4);           // buf0@kk done; buf1@kk+32 may be in flight
    __syncthreads();
    compute(0);
    __syncthreads();
    issue(0, kk + 64);
    WAIT_ASYNC(0x4);           // buf1@kk+32 done; buf0@kk+64 in flight
    __syncthreads();
    compute(1);
    __syncthreads();
    issue(1, kk + 96);
  }
  // tail: steps DMODEL-64 (buf0) and DMODEL-32 (buf1), nothing left to issue
  WAIT_ASYNC(0x4);
  __syncthreads();
  compute(0);
  __syncthreads();
  WAIT_ASYNC(0x0);
  __syncthreads();
  compute(1);

  // C layout: vgpr r -> M = r + 8*(lane/16), N = lane&15
#pragma unroll
  for (int ms = 0; ms < 2; ++ms) {
#pragma unroll
    for (int ns = 0; ns < 2; ++ns) {
      v8f g = ms ? (ns ? c111 : c110) : (ns ? c101 : c100);
      v8f u = ms ? (ns ? c311 : c310) : (ns ? c301 : c300);
      int n = n0 + wn * 32 + ns * 16 + l15;
      int mbase = wm * 32 + ms * 16 + half * 8;
#pragma unroll
      for (int r = 0; r < 8; ++r) {
        int m = mbase + r;
        if (m < valid) {
          float gv = g[r];
          float sil = gv / (1.0f + __expf(-gv));
          hb[(size_t)(row_start + m) * HDIM + n] = f2bfu(sil * u[r]);
        }
      }
    }
  }
}

// --------------------------------------------- GEMM2: out_perm = h @ w2 (f32)
__global__ __launch_bounds__(256) void moe_gemm2(
    const unsigned short* __restrict__ hb,
    const unsigned short* __restrict__ w2t,
    float* __restrict__ outp,
    const int* __restrict__ bspe) {
  __shared__ unsigned short As[2][128][40];
  __shared__ unsigned short Bs[2][64][40];

  const int tilesM = NKROWS / 128;
  int e  = blockIdx.x / tilesM;
  int mt = blockIdx.x - e * tilesM;
  int off = 0, gsz = 0;
#pragma unroll
  for (int j = 0; j < NEXP; ++j) { int b = bspe[j]; if (j < e) off += b; if (j == e) gsz = b; }
  if (mt * 128 >= gsz) return;
  int row_start = off + mt * 128;
  int valid = gsz - mt * 128; if (valid > 128) valid = 128;
  int n0 = blockIdx.y * 64;

  int t = threadIdx.x;
  int rowA = t >> 1;
  int chA  = (t & 1) * 16;
  int arow = row_start + rowA; if (arow >= NKROWS) arow = NKROWS - 1;
  const unsigned short* agA = hb + (size_t)arow * HDIM + chA;
  int rowB = t >> 2;
  int chB  = (t & 3) * 8;
  const unsigned short* bg = w2t + ((size_t)e << 20) + (size_t)(n0 + rowB) * HDIM + chB;

  int lane = t & 31;
  int wv   = t >> 5;
  int wm   = wv & 3, wn = wv >> 2;
  int half = lane >> 4;
  int l15  = lane & 15;
  int m0 = wm * 32 + l15, m1 = m0 + 16;
  int nA = wn * 32 + l15, nB = nA + 16;

  v8f c00 = {}, c01 = {}, c10 = {}, c11 = {};

  auto issue = [&](int buf, int kk) {
    async_ld_b128((unsigned)(size_t)&As[buf][rowA][chA],     agA + kk);
    async_ld_b128((unsigned)(size_t)&As[buf][rowA][chA + 8], agA + kk + 8);
    async_ld_b128((unsigned)(size_t)&Bs[buf][rowB][chB],     bg + kk);
  };

  auto compute = [&](int buf) {
    Frag a0, a1, f0, f1;
    a0.u[0] = *reinterpret_cast<const uint4v*>(&As[buf][m0][half * 8]);
    a0.u[1] = *reinterpret_cast<const uint4v*>(&As[buf][m0][half * 8 + 16]);
    a1.u[0] = *reinterpret_cast<const uint4v*>(&As[buf][m1][half * 8]);
    a1.u[1] = *reinterpret_cast<const uint4v*>(&As[buf][m1][half * 8 + 16]);
    f0.u[0] = *reinterpret_cast<const uint4v*>(&Bs[buf][nA][half * 8]);
    f0.u[1] = *reinterpret_cast<const uint4v*>(&Bs[buf][nA][half * 8 + 16]);
    f1.u[0] = *reinterpret_cast<const uint4v*>(&Bs[buf][nB][half * 8]);
    f1.u[1] = *reinterpret_cast<const uint4v*>(&Bs[buf][nB][half * 8 + 16]);

    c00 = __builtin_amdgcn_wmma_f32_16x16x32_bf16(false, a0.v, false, f0.v, (short)0, c00, false, false);
    c01 = __builtin_amdgcn_wmma_f32_16x16x32_bf16(false, a0.v, false, f1.v, (short)0, c01, false, false);
    c10 = __builtin_amdgcn_wmma_f32_16x16x32_bf16(false, a1.v, false, f0.v, (short)0, c10, false, false);
    c11 = __builtin_amdgcn_wmma_f32_16x16x32_bf16(false, a1.v, false, f1.v, (short)0, c11, false, false);
  };

  issue(0, 0);
  issue(1, 32);
  for (int kk = 0; kk + 96 < HDIM; kk += 64) {
    WAIT_ASYNC(0x3);
    __syncthreads();
    compute(0);
    __syncthreads();
    issue(0, kk + 64);
    WAIT_ASYNC(0x3);
    __syncthreads();
    compute(1);
    __syncthreads();
    issue(1, kk + 96);
  }
  WAIT_ASYNC(0x3);
  __syncthreads();
  compute(0);
  __syncthreads();
  WAIT_ASYNC(0x0);
  __syncthreads();
  compute(1);

#pragma unroll
  for (int ms = 0; ms < 2; ++ms) {
#pragma unroll
    for (int ns = 0; ns < 2; ++ns) {
      v8f c = ms ? (ns ? c11 : c10) : (ns ? c01 : c00);
      int n = n0 + wn * 32 + ns * 16 + l15;
      int mbase = wm * 32 + ms * 16 + half * 8;
#pragma unroll
      for (int r = 0; r < 8; ++r) {
        int m = mbase + r;
        if (m < valid) outp[(size_t)(row_start + m) * DMODEL + n] = c[r];
      }
    }
  }
}

// ------------------------------------------------- weighted top-k combine
__global__ __launch_bounds__(256) void moe_combine(
    const float* __restrict__ outp, const float* __restrict__ ew,
    const int* __restrict__ row_of, float* __restrict__ out) {
  int n = blockIdx.x;
  int t = threadIdx.x;                       // 256 * float4 = 1024
  int r0 = row_of[n * TOPK + 0];
  int r1 = row_of[n * TOPK + 1];
  float w0 = ew[n * TOPK + 0];
  float w1 = ew[n * TOPK + 1];
  float4v a = reinterpret_cast<const float4v*>(outp + (size_t)r0 * DMODEL)[t];
  float4v b = reinterpret_cast<const float4v*>(outp + (size_t)r1 * DMODEL)[t];
  float4v o = a * w0 + b * w1;
  reinterpret_cast<float4v*>(out + (size_t)n * DMODEL)[t] = o;
}

// ---------------------------------------------------------------------- launch
extern "C" void kernel_launch(void* const* d_in, const int* in_sizes, int n_in,
                              void* d_out, int out_size, void* d_ws, size_t ws_size,
                              hipStream_t stream) {
  const float* x    = (const float*)d_in[0];
  const float* ew   = (const float*)d_in[1];
  const int*   eidx = (const int*)d_in[2];
  const int*   bspe = (const int*)d_in[3];
  const float* w1   = (const float*)d_in[4];
  const float* w2   = (const float*)d_in[5];   // note dict order: w1, w2, w3
  const float* w3   = (const float*)d_in[6];
  float* out = (float*)d_out;

  char* ws = (char*)d_ws;
  size_t o = 0;
  int* cnt    = (int*)(ws + o); o += 256;
  int* row_of = (int*)(ws + o); o += (size_t)NKROWS * 4;
  int* src_of = (int*)(ws + o); o += (size_t)NKROWS * 4;
  unsigned short* xb  = (unsigned short*)(ws + o); o += (size_t)NKROWS * DMODEL * 2;
  unsigned short* w1t = (unsigned short*)(ws + o); o += (size_t)NEXP * DMODEL * HDIM * 2;
  unsigned short* w3t = (unsigned short*)(ws + o); o += (size_t)NEXP * DMODEL * HDIM * 2;
  unsigned short* w2t = (unsigned short*)(ws + o); o += (size_t)NEXP * HDIM * DMODEL * 2;
  unsigned short* hb  = (unsigned short*)(ws + o); o += (size_t)NKROWS * HDIM * 2;
  float* outp = (float*)(ws + o);

  moe_zero<<<1, 32, 0, stream>>>(cnt);
  moe_perm<<<NKROWS / 256, 256, 0, stream>>>(eidx, bspe, cnt, row_of, src_of);
  moe_wcvt<<<dim3(32, 32, 3 * NEXP), 256, 0, stream>>>(w1, w2, w3, w1t, w2t, w3t);
  moe_gather_x<<<NKROWS, 256, 0, stream>>>(x, src_of, xb);

  dim3 g1(NEXP * (NKROWS / 128), HDIM / 64);
  moe_gemm1<<<g1, 256, 0, stream>>>(xb, w1t, w3t, hb, bspe);
  dim3 g2(NEXP * (NKROWS / 128), DMODEL / 64);
  moe_gemm2<<<g2, 256, 0, stream>>>(hb, w2t, outp, bspe);

  moe_combine<<<N_TOK, 256, 0, stream>>>(outp, ew, row_of, out);
}
